// Reservoir_82274393522504
// MI455X (gfx1250) — matrix operand order repaired and checked by reading, not compiled, not run on hardware
//
#include <hip/hip_runtime.h>
#include <hip/hip_bf16.h>
#include <math.h>

typedef __attribute__((ext_vector_type(8)))  __bf16 v8bf;
typedef __attribute__((ext_vector_type(16))) __bf16 v16bf;
typedef __attribute__((ext_vector_type(8)))  float  v8f;
typedef __attribute__((ext_vector_type(4)))  int    v4i;

// Address-space-qualified pointer types for the async-to-LDS builtin:
// param0 = v4i in global (AS1), param1 = v4i in LDS (AS3).
typedef __attribute__((address_space(1))) v4i* as1_v4i_ptr;
typedef __attribute__((address_space(3))) v4i* as3_v4i_ptr;

#define DN     4160   // D + N
#define DHEAD  64     // D
#define BATCH  256    // B
#define TSTEPS 100    // T

#define KDEPTH 64     // K-panel depth staged per barrier (2 WMMA k-chunks)
#define LROW   72     // LDS row stride (elems): 16B-aligned, bank-conflict-free

#if defined(__has_builtin)
#if __has_builtin(__builtin_amdgcn_global_load_async_to_lds_b128)
#define USE_ASYNC_LDS 1
#endif
#endif

// Stage one 16-byte chunk global -> LDS (per thread).
__device__ __forceinline__ void stage16(const __bf16* g, __bf16* l) {
#if defined(USE_ASYNC_LDS)
  __builtin_amdgcn_global_load_async_to_lds_b128(
      (as1_v4i_ptr)(g), (as3_v4i_ptr)(l), 0, 0);
#else
  *(v8bf*)l = *(const v8bf*)g;   // sync fallback: global load + ds_store
#endif
}

__device__ __forceinline__ void stage_wait() {
#if defined(USE_ASYNC_LDS)
#if __has_builtin(__builtin_amdgcn_s_wait_asynccnt)
  __builtin_amdgcn_s_wait_asynccnt(0);
#else
  asm volatile("s_wait_asynccnt 0x0" ::: "memory");
#endif
#endif
}

// A fragment (16x32 A, contiguous k = 16*hi + s order): two contiguous 16B chunks.
__device__ __forceinline__ v16bf loadA(const __bf16* arow, int kk) {
  v8bf lo = *(const v8bf*)(arow + kk);
  v8bf hi = *(const v8bf*)(arow + kk + 8);
  return __builtin_shufflevector(lo, hi, 0,1,2,3,4,5,6,7,8,9,10,11,12,13,14,15);
}

// B fragment from the LDS panel: row = 16j + l15, k-subchunk s (0/1),
// 16 contiguous K values at s*32 + hi*16.
__device__ __forceinline__ v16bf ldsB(const __bf16* panel, int j, int s,
                                      int l15, int hi) {
  const __bf16* p = panel + (16 * j + l15) * LROW + s * 32 + hi * 16;
  v8bf lo = *(const v8bf*)p;
  v8bf hi8 = *(const v8bf*)(p + 8);
  return __builtin_shufflevector(lo, hi8, 0,1,2,3,4,5,6,7,8,9,10,11,12,13,14,15);
}

// ---------------------------------------------------------------------------
// Prologue kernels
// ---------------------------------------------------------------------------
__global__ void f32_to_bf16_kernel(const float* __restrict__ src,
                                   __bf16* __restrict__ dst, int n) {
  int i = blockIdx.x * blockDim.x + threadIdx.x;
  if (i < n) dst[i] = (__bf16)src[i];
}

__global__ void transpose_f32_to_bf16(const float* __restrict__ src,
                                      __bf16* __restrict__ dst,
                                      int rows, int cols) {
  __shared__ float tile[32][33];
  const int c0 = blockIdx.x * 32;
  const int r0 = blockIdx.y * 32;
  const int tx = threadIdx.x;      // 0..31
  const int ty = threadIdx.y;      // 0..7
#pragma unroll
  for (int j = 0; j < 32; j += 8) {
    int r = r0 + ty + j, c = c0 + tx;
    if (r < rows && c < cols) tile[ty + j][tx] = src[(size_t)r * cols + c];
  }
  __syncthreads();
#pragma unroll
  for (int j = 0; j < 32; j += 8) {
    int r = c0 + ty + j, c = r0 + tx;          // dst is cols x rows
    if (r < cols && c < rows) dst[(size_t)r * rows + c] = (__bf16)tile[tx][ty + j];
  }
}

// ---------------------------------------------------------------------------
// GEMM1 (fused): R = act( X(256x4160) @ K(4160x4160) + b )
// 8 waves / WG share one 64-col N block; a 64(N) x 64(K) bf16 B panel (8KB)
// is staged once per WG into LDS via async-to-LDS, double-buffered. Each
// barrier pair now covers 8 WMMAs (2 k-chunks x 4 accumulators).
// Grid: (DN/64, BATCH/128). Head (cols < 64) is exactly blockIdx.x==0.
// ---------------------------------------------------------------------------
__global__ __launch_bounds__(256) void reservoir_gemm1(
    const __bf16* __restrict__ X, const __bf16* __restrict__ KT,
    const float* __restrict__ bias, __bf16* __restrict__ R) {
  __shared__ __align__(16) __bf16 sbuf[2][64 * LROW];

  const int lane = threadIdx.x & 31;
  const int wave = threadIdx.x >> 5;
  const int hi   = lane >> 4;
  const int l15  = lane & 15;
  const int m0   = blockIdx.y * 128 + wave * 16;
  const int n0   = blockIdx.x * 64;

  // Cooperative staging: 64 rows x 64 elems = 512 16B-chunks; each of the
  // 256 threads copies 2 chunks (rows tid/8 and 32 + tid/8).
  const int tid   = threadIdx.x;
  const int row0  = tid >> 3;
  const int row1  = row0 + 32;
  const int koff  = (tid & 7) * 8;
  const __bf16* g0 = KT + (size_t)(n0 + row0) * DN + koff;
  const __bf16* g1 = KT + (size_t)(n0 + row1) * DN + koff;
  __bf16* l0[2] = { &sbuf[0][row0 * LROW + koff], &sbuf[1][row0 * LROW + koff] };
  __bf16* l1[2] = { &sbuf[0][row1 * LROW + koff], &sbuf[1][row1 * LROW + koff] };

  stage16(g0, l0[0]);                // preload panel kk=0 into buffer 0
  stage16(g1, l1[0]);
  stage_wait();
  __syncthreads();

  const __bf16* arow = X + (size_t)(m0 + l15) * DN + hi * 16;
  v16bf a0 = loadA(arow, 0);         // A fragments pipelined in registers
  v16bf a1 = loadA(arow, 32);

  v8f acc[4] = {v8f{}, v8f{}, v8f{}, v8f{}};

  for (int kk = 0, it = 0; kk < DN; kk += KDEPTH, it ^= 1) {
    const __bf16* panel = sbuf[it];
    const int kkn = (kk + KDEPTH < DN) ? (kk + KDEPTH) : 0;  // wrap: keep legal

    stage16(g0 + kkn, l0[it ^ 1]);   // async: next panel -> other buffer
    stage16(g1 + kkn, l1[it ^ 1]);

    v16bf a_use0 = a0;
    v16bf a_use1 = a1;
    a0 = loadA(arow, kkn);           // prefetch next A into registers
    a1 = loadA(arow, kkn + 32);

#pragma unroll
    for (int s = 0; s < 2; ++s) {
      v16bf av = s ? a_use1 : a_use0;
      v16bf b0 = ldsB(panel, 0, s, l15, hi);
      v16bf b1 = ldsB(panel, 1, s, l15, hi);
      v16bf b2 = ldsB(panel, 2, s, l15, hi);
      v16bf b3 = ldsB(panel, 3, s, l15, hi);
      acc[0] = __builtin_amdgcn_wmma_f32_16x16x32_bf16(false, av, false, b0,
                                                       (short)0, acc[0], false, false);
      acc[1] = __builtin_amdgcn_wmma_f32_16x16x32_bf16(false, av, false, b1,
                                                       (short)0, acc[1], false, false);
      acc[2] = __builtin_amdgcn_wmma_f32_16x16x32_bf16(false, av, false, b2,
                                                       (short)0, acc[2], false, false);
      acc[3] = __builtin_amdgcn_wmma_f32_16x16x32_bf16(false, av, false, b3,
                                                       (short)0, acc[3], false, false);
    }

    stage_wait();        // next panel landed in LDS
    __syncthreads();     // publish it; also proves all waves done reading 'panel'
  }

  const bool head = (blockIdx.x == 0);
#pragma unroll
  for (int j = 0; j < 4; ++j) {
    const int col = n0 + j * 16 + l15;
    const float bc = bias[col];
#pragma unroll
    for (int g = 0; g < 8; ++g) {        // D layout: M = m0 + g + hi*8
      const int row = m0 + g + hi * 8;
      float z = acc[j][g] + bc;
      float r = head ? z : (0.1f * z + 0.9f * tanhf(z));
      R[(size_t)row * DN + col] = (__bf16)r;
    }
  }
}

// ---------------------------------------------------------------------------
// GEMM2 (fused): u = R @ Wout + bout; writes fp32 out slice for step t and
// overwrites R[:, :64] with bf16(u) so R becomes the next state in place.
// ---------------------------------------------------------------------------
__global__ __launch_bounds__(256) void reservoir_gemm2(
    __bf16* __restrict__ R, const __bf16* __restrict__ WT,
    const float* __restrict__ bout, float* __restrict__ out, int t) {
  const int lane = threadIdx.x & 31;
  const int wave = threadIdx.x >> 5;
  const int hi   = lane >> 4;
  const int l15  = lane & 15;
  const int m0   = blockIdx.y * 128 + wave * 16;

  const __bf16* arow = R + (size_t)(m0 + l15) * DN + hi * 16;
  const __bf16* b0 = WT + (size_t)l15 * DN + hi * 16;   // WT: 64 x 4160
  const __bf16* b1 = b0 + (size_t)16 * DN;
  const __bf16* b2 = b0 + (size_t)32 * DN;
  const __bf16* b3 = b0 + (size_t)48 * DN;

  v8f acc[4] = {v8f{}, v8f{}, v8f{}, v8f{}};

  v16bf a = loadA(arow, 0);
  for (int kk = 0; kk < DN; kk += 32) {
    const int kkn = (kk + 32 < DN) ? (kk + 32) : 0;
    v16bf bb0 = *(const v16bf*)(b0 + kk);
    v16bf bb1 = *(const v16bf*)(b1 + kk);
    v16bf bb2 = *(const v16bf*)(b2 + kk);
    v16bf bb3 = *(const v16bf*)(b3 + kk);
    v16bf a_use = a;
    a = loadA(arow, kkn);
    acc[0] = __builtin_amdgcn_wmma_f32_16x16x32_bf16(false, a_use, false, bb0,
                                                     (short)0, acc[0], false, false);
    acc[1] = __builtin_amdgcn_wmma_f32_16x16x32_bf16(false, a_use, false, bb1,
                                                     (short)0, acc[1], false, false);
    acc[2] = __builtin_amdgcn_wmma_f32_16x16x32_bf16(false, a_use, false, bb2,
                                                     (short)0, acc[2], false, false);
    acc[3] = __builtin_amdgcn_wmma_f32_16x16x32_bf16(false, a_use, false, bb3,
                                                     (short)0, acc[3], false, false);
  }

#pragma unroll
  for (int j = 0; j < 4; ++j) {
    const int col = j * 16 + l15;           // 0..63
    const float bc = bout[col];
#pragma unroll
    for (int g = 0; g < 8; ++g) {
      const int row = m0 + g + hi * 8;
      const float u = acc[j][g] + bc;
      out[(size_t)row * (TSTEPS * DHEAD) + (size_t)t * DHEAD + col] = u;
      R[(size_t)row * DN + col] = (__bf16)u;   // state head for next step
    }
  }
}

// ---------------------------------------------------------------------------
// Host side
// ---------------------------------------------------------------------------
extern "C" void kernel_launch(void* const* d_in, const int* in_sizes, int n_in,
                              void* d_out, int out_size, void* d_ws, size_t ws_size,
                              hipStream_t stream) {
  const float* inputs = (const float*)d_in[0];   // 256 x 4160
  const float* Kmat   = (const float*)d_in[1];   // 4160 x 4160
  const float* bvec   = (const float*)d_in[2];   // 4160
  const float* Wout   = (const float*)d_in[3];   // 4160 x 64
  const float* bout   = (const float*)d_in[4];   // 64
  // d_in[5] = num_steps (device scalar); T=100 fixed by the problem config.

  char* ws = (char*)d_ws;
  size_t off = 0;
  auto take = [&](size_t bytes) -> char* {
    char* p = ws + off;
    off += (bytes + 255) & ~(size_t)255;
    return p;
  };
  __bf16* KT = (__bf16*)take((size_t)DN * DN * sizeof(__bf16));       // 34.6 MB
  __bf16* WT = (__bf16*)take((size_t)DHEAD * DN * sizeof(__bf16));    // 0.53 MB
  __bf16* Xa = (__bf16*)take((size_t)BATCH * DN * sizeof(__bf16));    // 2.13 MB
  __bf16* Xb = (__bf16*)take((size_t)BATCH * DN * sizeof(__bf16));    // 2.13 MB

  transpose_f32_to_bf16<<<dim3(DN / 32, DN / 32), dim3(32, 8), 0, stream>>>(
      Kmat, KT, DN, DN);
  transpose_f32_to_bf16<<<dim3(DHEAD / 32, DN / 32), dim3(32, 8), 0, stream>>>(
      Wout, WT, DN, DHEAD);
  f32_to_bf16_kernel<<<(BATCH * DN + 255) / 256, 256, 0, stream>>>(
      inputs, Xa, BATCH * DN);

  __bf16* X = Xa;
  __bf16* R = Xb;
  for (int t = 0; t < TSTEPS; ++t) {
    reservoir_gemm1<<<dim3(DN / 64, BATCH / 128), 256, 0, stream>>>(X, KT, bvec, R);
    reservoir_gemm2<<<dim3(1, BATCH / 128), 256, 0, stream>>>(R, WT, bout,
                                                              (float*)d_out, t);
    __bf16* tmp = X; X = R; R = tmp;   // R (head replaced by u) is next state
  }
}
